// MultiTimescaleRNN_59828894433410
// MI455X (gfx1250) — matrix operand order repaired and checked by reading, not compiled, not run on hardware
//
#include <hip/hip_runtime.h>
#include <math.h>

#define H  2048
#define O  512
#define B  64
#define T  512
#define SCAN_GRID 64   // 64 workgroups x 8 waves = 512 waves = 4 x 128 tiles

typedef __attribute__((ext_vector_type(16))) __bf16 v16bf;
typedef __attribute__((ext_vector_type(8)))  __bf16 v8bf;
typedef __attribute__((ext_vector_type(8)))  float  v8f;

// Branch-free tanh on HW trans ops: tanh(x) = sign(x) * (1 - 2/(e^{2|x|}+1))
__device__ __forceinline__ float fast_tanh(float x) {
  float ax = __builtin_fabsf(x);
  float e  = __builtin_amdgcn_exp2f(2.885390082f * ax);   // e^{2|x|}
  float r  = 1.0f - 2.0f * __builtin_amdgcn_rcpf(e + 1.0f);
  return __builtin_copysignf(r, x);
}

// ---------------------------------------------------------------------------
// Pack W (n_total x H row-major f32, used as y = x @ W^T) into WMMA-B fragment
// order, bf16. e = ((nt*64 + kb)*32 + lane)*16 + j ;
// lane n = nt*16 + (l&15), k = kb*32 + (j>>3)*16 + (l>>4)*8 + (j&7)  (ISA 7.12.2, wave32).
// ---------------------------------------------------------------------------
__global__ void pack_wT_bf16(const float* __restrict__ W, __bf16* __restrict__ P,
                             int n_total) {
  int e = blockIdx.x * blockDim.x + threadIdx.x;
  if (e >= n_total * H) return;
  int j  = e & 15;
  int l  = (e >> 4) & 31;
  int kb = (e >> 9) & 63;
  int nt = e >> 15;
  int k  = kb * 32 + ((j >> 3) * 16) + ((l >> 4) * 8) + (j & 7);
  int n  = nt * 16 + (l & 15);
  P[e] = (__bf16)W[n * H + k];
}

// ---------------------------------------------------------------------------
// h0 = place_cells_0 @ W_h_init^T : (64,512)@(512,2048) -> f32 + bf16 copies
// ---------------------------------------------------------------------------
__global__ void init_h0(const float* __restrict__ pc, const float* __restrict__ Wh,
                        float* __restrict__ hf, __bf16* __restrict__ hbf) {
  int idx = blockIdx.x * blockDim.x + threadIdx.x;
  if (idx >= B * H) return;
  int m = idx / H, n = idx - m * H;
  float s = 0.f;
  for (int k = 0; k < O; ++k) s = fmaf(pc[m * O + k], Wh[n * O + k], s);
  hf[idx]  = s;
  hbf[idx] = (__bf16)s;
}

__global__ void zero_ctr(unsigned* ctr) { *ctr = 0u; }

// ---------------------------------------------------------------------------
// Persistent scan: ALL 512 timesteps in one launch. 64 blocks x 8 waves; wave
// (mt,nt) owns a 16x16 tile of h for the whole scan; h-f32 lives in VGPRs.
// Cross-step sync: monotonic device-scope atomic counter barrier (64 blocks
// are trivially co-resident on MI455X).
// ---------------------------------------------------------------------------
__global__ __launch_bounds__(256)
void rnn_scan(const __bf16* __restrict__ wrec_pk,
              __bf16* __restrict__ hbf0, __bf16* __restrict__ hbf1,
              const float* __restrict__ hf_init,
              const float* __restrict__ inp,  const float* __restrict__ Wi,
              const float* __restrict__ bi,   const float* __restrict__ br,
              const float* __restrict__ alphas,
              float* __restrict__ hiddens, unsigned* __restrict__ ctr) {
  int wid = blockIdx.x * 8 + (threadIdx.x >> 5);
  int l   = threadIdx.x & 31;
  int mt  = wid >> 7;      // 0..3   (batch tile)
  int nt  = wid & 127;     // 0..127 (hidden tile)

  // per-lane constants, hoisted for all 512 steps
  int n      = nt * 16 + (l & 15);
  float al   = alphas[n];
  float bias = bi[n] + br[n];
  float wi0  = Wi[n * 2 + 0];
  float wi1  = Wi[n * 2 + 1];
  int mbase  = mt * 16 + (l >> 4) * 8;

  // resident f32 h tile: only this wave ever touches these (m,n) positions
  float hreg[8];
#pragma unroll
  for (int v = 0; v < 8; ++v)
    hreg[v] = hf_init[(size_t)(mbase + v) * H + n];

  size_t aoff = (size_t)(mt * 16 + (l & 15)) * H + ((l >> 4) * 8);
  const __bf16* bbase = wrec_pk + ((size_t)nt * 64 * 32 + l) * 16;

  for (int t = 0; t < T; ++t) {
    const __bf16* acur = ((t & 1) ? hbf1 : hbf0) + aoff;
    __bf16*       hnxt = ((t & 1) ? hbf0 : hbf1);

    v8f acc = {};
#pragma unroll 4
    for (int kb = 0; kb < 64; ++kb) {
      v8bf alo = *(const v8bf*)(acur + kb * 32);        // K = k0 .. k0+7
      v8bf ahi = *(const v8bf*)(acur + kb * 32 + 16);   // K = k0+16 .. k0+23
      v16bf a = __builtin_shufflevector(alo, ahi,
                  0,1,2,3,4,5,6,7,8,9,10,11,12,13,14,15);
      v16bf b = *(const v16bf*)(bbase + (size_t)kb * 512);
      acc = __builtin_amdgcn_wmma_f32_16x16x32_bf16(false, a, false, b,
                                                    (short)0, acc, false, false);
    }

    // epilogue: C/D layout -> lane n = l&15, m = (l>>4)*8 + v
#pragma unroll
    for (int v = 0; v < 8; ++v) {
      int m     = mbase + v;
      float2 x  = *(const float2*)(inp + ((size_t)m * T + t) * 2);
      float pre = acc[v] + fmaf(x.x, wi0, fmaf(x.y, wi1, bias));
      float act = fast_tanh(pre);
      float hn  = fmaf(al, act - hreg[v], hreg[v]);
      hreg[v]   = hn;
      hnxt[(size_t)m * H + n]              = (__bf16)hn;
      hiddens[((size_t)m * T + t) * H + n] = hn;       // hiddens[b,t,h]
    }

    // ---- grid barrier (release -> arrive -> wait -> acquire) ----
    __threadfence();           // make hnxt stores device-visible
    __syncthreads();
    if (threadIdx.x == 0) {
      unsigned target = (unsigned)(t + 1) * SCAN_GRID;  // monotonic, no reset
      __hip_atomic_fetch_add(ctr, 1u, __ATOMIC_RELEASE, __HIP_MEMORY_SCOPE_AGENT);
      while (__hip_atomic_load(ctr, __ATOMIC_RELAXED, __HIP_MEMORY_SCOPE_AGENT) < target)
        __builtin_amdgcn_s_sleep(1);
    }
    __syncthreads();
    __threadfence();           // acquire: invalidate near caches before next A reads
  }
}

// ---------------------------------------------------------------------------
// outputs = hiddens @ W_out^T : (B*T, H) @ (H, O). A converted f32->bf16 on
// the fly; each wave computes a 16(M) x 64(N) stripe (4 N-tiles share one A
// fragment). 2048 m-tiles x 8 n-groups = 16384 waves.
// ---------------------------------------------------------------------------
__global__ __launch_bounds__(256)
void out_gemm(const float* __restrict__ hid, const __bf16* __restrict__ wout_pk,
              float* __restrict__ out) {
  int wid = blockIdx.x * 8 + (threadIdx.x >> 5);   // 0..16383
  int l   = threadIdx.x & 31;
  int mt  = wid >> 3;     // 0..2047
  int ng  = wid & 7;      // n-group of 4 tiles

  const float*  arow = hid + (size_t)(mt * 16 + (l & 15)) * H + ((l >> 4) * 8);
  const __bf16* bl   = wout_pk + ((size_t)(4 * ng) * 64 * 32 + l) * 16;

  v8f acc0 = {}, acc1 = {}, acc2 = {}, acc3 = {};
  for (int kb = 0; kb < 64; ++kb) {
    const float* p = arow + kb * 32;
    float4 f0 = *(const float4*)(p);
    float4 f1 = *(const float4*)(p + 4);
    float4 f2 = *(const float4*)(p + 16);
    float4 f3 = *(const float4*)(p + 20);
    v16bf a;
    a[0]=(__bf16)f0.x;  a[1]=(__bf16)f0.y;  a[2]=(__bf16)f0.z;  a[3]=(__bf16)f0.w;
    a[4]=(__bf16)f1.x;  a[5]=(__bf16)f1.y;  a[6]=(__bf16)f1.z;  a[7]=(__bf16)f1.w;
    a[8]=(__bf16)f2.x;  a[9]=(__bf16)f2.y;  a[10]=(__bf16)f2.z; a[11]=(__bf16)f2.w;
    a[12]=(__bf16)f3.x; a[13]=(__bf16)f3.y; a[14]=(__bf16)f3.z; a[15]=(__bf16)f3.w;
    v16bf b0 = *(const v16bf*)(bl + (size_t)kb * 512);
    v16bf b1 = *(const v16bf*)(bl + (size_t)(64 * 32 * 16) + (size_t)kb * 512);
    v16bf b2 = *(const v16bf*)(bl + (size_t)(2 * 64 * 32 * 16) + (size_t)kb * 512);
    v16bf b3 = *(const v16bf*)(bl + (size_t)(3 * 64 * 32 * 16) + (size_t)kb * 512);
    acc0 = __builtin_amdgcn_wmma_f32_16x16x32_bf16(false, a, false, b0, (short)0, acc0, false, false);
    acc1 = __builtin_amdgcn_wmma_f32_16x16x32_bf16(false, a, false, b1, (short)0, acc1, false, false);
    acc2 = __builtin_amdgcn_wmma_f32_16x16x32_bf16(false, a, false, b2, (short)0, acc2, false, false);
    acc3 = __builtin_amdgcn_wmma_f32_16x16x32_bf16(false, a, false, b3, (short)0, acc3, false, false);
  }
  int nb    = (4 * ng) * 16 + (l & 15);
  int mbase = mt * 16 + (l >> 4) * 8;
#pragma unroll
  for (int v = 0; v < 8; ++v) {
    size_t row = (size_t)(mbase + v) * O;
    out[row + nb +  0] = acc0[v];
    out[row + nb + 16] = acc1[v];
    out[row + nb + 32] = acc2[v];
    out[row + nb + 48] = acc3[v];
  }
}

// ---------------------------------------------------------------------------
extern "C" void kernel_launch(void* const* d_in, const int* in_sizes, int n_in,
                              void* d_out, int out_size, void* d_ws, size_t ws_size,
                              hipStream_t stream) {
  const float* inputs = (const float*)d_in[0];  // (B,T,2)
  const float* pc0    = (const float*)d_in[1];  // (B,O)
  const float* Wi     = (const float*)d_in[2];  // (H,2)
  const float* bi     = (const float*)d_in[3];  // (H,)
  const float* Wr     = (const float*)d_in[4];  // (H,H)
  const float* br     = (const float*)d_in[5];  // (H,)
  const float* Wo     = (const float*)d_in[6];  // (O,H)
  const float* Wh     = (const float*)d_in[7];  // (H,O)
  const float* al     = (const float*)d_in[8];  // (H,)

  float* outputs = (float*)d_out;                    // (B,T,O)
  float* hiddens = outputs + (size_t)B * T * O;      // (B,T,H)

  char* ws = (char*)d_ws;
  __bf16*  wrec_pk = (__bf16*)ws;                                  // 8 MB
  __bf16*  wout_pk = (__bf16*)(ws + (size_t)H * H * 2);            // 2 MB
  float*   hf0     = (float*)(ws + (size_t)H * H * 2 + (size_t)O * H * 2);
  __bf16*  hbf0    = (__bf16*)(hf0 + (size_t)B * H);
  __bf16*  hbf1    = hbf0 + (size_t)B * H;
  unsigned* ctr    = (unsigned*)(hbf1 + (size_t)B * H);

  // 1) pack weights into WMMA-B fragment order (bf16); zero barrier counter
  pack_wT_bf16<<<(H * H + 255) / 256, 256, 0, stream>>>(Wr, wrec_pk, H);
  pack_wT_bf16<<<(O * H + 255) / 256, 256, 0, stream>>>(Wo, wout_pk, O);
  zero_ctr<<<1, 1, 0, stream>>>(ctr);

  // 2) h0 from place cells (f32 for register preload, bf16 for step-0 GEMM)
  init_h0<<<(B * H + 255) / 256, 256, 0, stream>>>(pc0, Wh, hf0, hbf0);

  // 3) entire 512-step scan in ONE persistent launch (grid-barrier per step)
  rnn_scan<<<SCAN_GRID, 256, 0, stream>>>(wrec_pk, hbf0, hbf1, hf0,
                                          inputs, Wi, bi, br, al,
                                          hiddens, ctr);

  // 4) output projection over all (b,t)
  out_gemm<<<2048, 256, 0, stream>>>(hiddens, wout_pk, outputs);
}
